// SNA_1082331759234
// MI455X (gfx1250) — matrix operand order repaired and checked by reading, not compiled, not run on hardware
//
#include <hip/hip_runtime.h>
#include <hip/hip_bf16.h>

// ---------------- problem constants ----------------
#define NN 50000
#define TT 4
#define KK 16
#define UC 64
#define CC 128
#define C4 32
#define NT (NN * TT)        // 200000 rows
#define TILES (NT / 16)     // 12500 16-row WMMA tiles
#define WAVES 8             // waves per block in tile kernels
#define LDSW_BF 136         // bf16 LDS row stride for 16x128 activation tiles
#define IN_LDSW 68          // bf16 LDS row stride for 16x64 input tiles
#define BN_EPS 1e-5f

// weight-fragment table (each fragment = 32 lanes x 16 bf16 = 1KB)
#define FRAG_LIN 0          // 16 frags: kb{0,1} x nt{0..7}, row-major W_lin
#define FRAG_QK 16          //  8 frags: kb{0..3} x nt{0,1}, W_qk^T
#define FRAG_V 24           // 32 frags: kb{0..3} x nt{0..7}, W_v^T
#define FRAG_T 56           // 32 frags: kb{0..3} x nt{0..7}, W_t^T
#define NFRAG 88

// ---------------- WMMA types / helpers ----------------
typedef __bf16 bf16;
typedef bf16         v16bf __attribute__((ext_vector_type(16)));
typedef float        v8f   __attribute__((ext_vector_type(8)));
typedef unsigned int v8u   __attribute__((ext_vector_type(8)));

// Let the backend pick the hardware bf16 convert (fptrunc is always legalizable).
__device__ __forceinline__ bf16 f2bf(float f) { return (bf16)f; }

__device__ __forceinline__ unsigned int pack2bf(float lo, float hi) {
    unsigned short a = __builtin_bit_cast(unsigned short, (bf16)lo);
    unsigned short b = __builtin_bit_cast(unsigned short, (bf16)hi);
    return (unsigned int)a | ((unsigned int)b << 16);
}

__device__ __forceinline__ v8f wmma_bf16(v16bf a, v16bf b, v8f c) {
    return __builtin_amdgcn_wmma_f32_16x16x32_bf16(false, a, false, b, (short)0, c,
                                                   false, false);
}

// A fragment from a bf16 LDS tile (row stride ld): contiguous K-pairs -> 8x ds_load_b32.
__device__ __forceinline__ v16bf load_A_ldsbf(const bf16* t, int ld, int kbase, int lane) {
    int m = lane & 15, half = lane >> 4;
    const bf16* row = t + m * ld + kbase + half * 8;
    v8u u;
#pragma unroll
    for (int p = 0; p < 8; ++p) {
        int base = (p < 4) ? (p * 2) : (16 + (p - 4) * 2);
        u[p] = *(const unsigned int*)(row + base);
    }
    return __builtin_bit_cast(v16bf, u);
}

// Prepacked weight fragment: one 32B vector load (2x global_load_b128).
__device__ __forceinline__ v16bf ldfrag(const bf16* __restrict__ wfrag, int f, int lane) {
    return *(const v16bf*)(wfrag + ((size_t)f * 32 + lane) * 16);
}

// B[k][n] = W[k*ld + n]   (row-major K x N weight) -- pack kernel only
__device__ __forceinline__ v16bf load_B_rm(const float* __restrict__ W, int ld,
                                           int kbase, int n0, int lane) {
    int n = n0 + (lane & 15), half = lane >> 4;
    v16bf b;
#pragma unroll
    for (int e = 0; e < 16; ++e)
        b[e] = f2bf(W[(kbase + ((e < 8) ? (half * 8 + e) : (16 + half * 8 + e - 8))) * ld + n]);
    return b;
}

// B[k][n] = W[n*ld + k]   (weight stored (Nout, K)) -- pack kernel only
__device__ __forceinline__ v16bf load_B_tr(const float* __restrict__ W, int ld,
                                           int kbase, int n0, int lane) {
    int n = n0 + (lane & 15), half = lane >> 4;
    v16bf b;
#pragma unroll
    for (int e = 0; e < 16; ++e)
        b[e] = f2bf(W[n * ld + kbase + ((e < 8) ? (half * 8 + e) : (16 + half * 8 + e - 8))]);
    return b;
}

// ---------------- kernel 0: zero accumulators ----------------
__global__ void sna_zero(float* __restrict__ p, int n) {
    int i = blockIdx.x * blockDim.x + threadIdx.x;
    if (i < n) p[i] = 0.f;
}

// ---------------- kernel 0b: pack weights into bf16 WMMA fragments ----------------
__global__ void __launch_bounds__(32)
sna_pack(const float* __restrict__ W_lin, const float* __restrict__ W_qk,
         const float* __restrict__ W_v, const float* __restrict__ W_t,
         bf16* __restrict__ wfrag) {
    int f = blockIdx.x, lane = threadIdx.x;
    v16bf b;
    if (f < FRAG_QK) {
        b = load_B_rm(W_lin, CC, (f >> 3) * 32, (f & 7) * 16, lane);
    } else if (f < FRAG_V) {
        int g = f - FRAG_QK;
        b = load_B_tr(W_qk, CC, (g >> 1) * 32, (g & 1) * 16, lane);
    } else if (f < FRAG_T) {
        int g = f - FRAG_V;
        b = load_B_tr(W_v, CC, (g >> 3) * 32, (g & 7) * 16, lane);
    } else {
        int g = f - FRAG_T;
        b = load_B_tr(W_t, CC, (g >> 3) * 32, (g & 7) * 16, lane);
    }
    *(v16bf*)(wfrag + ((size_t)f * 32 + lane) * 16) = b;
}

// ---------------- kernel 1: fused linear stage ----------------
template <bool IS_X>
__device__ __forceinline__ void
linear_tile(const float* __restrict__ src, const bf16* __restrict__ wfrag,
            const float* __restrict__ b_lin, const float* __restrict__ b_v,
            float* __restrict__ xf, float* __restrict__ qdst, float* __restrict__ yv,
            bf16* itile, bf16* tile, int rowbase, int lane) {
    const int m15 = lane & 15, half = lane >> 4;
    // ---- stage contiguous 16x64 f32 input block -> bf16 LDS (coalesced b128) ----
    {
        const float4* srcv = (const float4*)(src + (size_t)rowbase * UC);
#pragma unroll
        for (int i = 0; i < 8; ++i) {
            int idx4 = i * 32 + lane;          // 256 float4s in the block
            int rr = idx4 >> 4;                // row (16 float4 per row)
            int cc = (idx4 & 15) << 2;         // column (floats)
            float4 v = srcv[idx4];
            *(unsigned int*)(itile + rr * IN_LDSW + cc)     = pack2bf(v.x, v.y);
            *(unsigned int*)(itile + rr * IN_LDSW + cc + 2) = pack2bf(v.z, v.w);
        }
    }
    v16bf a0 = load_A_ldsbf(itile, IN_LDSW, 0, lane);
    v16bf a1 = load_A_ldsbf(itile, IN_LDSW, 32, lane);
    // ---- GEMM1: (16 x 64) @ (64 x 128) ----
#pragma unroll 2
    for (int nt = 0; nt < 8; ++nt) {
        v8f acc = {};
        acc = wmma_bf16(a0, ldfrag(wfrag, FRAG_LIN + nt, lane), acc);
        acc = wmma_bf16(a1, ldfrag(wfrag, FRAG_LIN + 8 + nt, lane), acc);
        float bias = b_lin[nt * 16 + m15];
#pragma unroll
        for (int r = 0; r < 8; ++r) {
            int mm = r + half * 8;
            float v = acc[r] + bias;
            tile[mm * LDSW_BF + nt * 16 + m15] = f2bf(v);
            if constexpr (IS_X)
                xf[(size_t)(rowbase + mm) * CC + nt * 16 + m15] = v;
        }
    }
    // ---- A fragments of the activation tile ----
    v16bf af[4];
#pragma unroll
    for (int kb = 0; kb < 4; ++kb) af[kb] = load_A_ldsbf(tile, LDSW_BF, kb * 32, lane);
    // ---- q/k projection: (16 x 128) @ (128 x 32) ----
#pragma unroll
    for (int nt = 0; nt < 2; ++nt) {
        v8f acc = {};
#pragma unroll
        for (int kb = 0; kb < 4; ++kb)
            acc = wmma_bf16(af[kb], ldfrag(wfrag, FRAG_QK + kb * 2 + nt, lane), acc);
#pragma unroll
        for (int r = 0; r < 8; ++r)
            qdst[(size_t)(rowbase + r + half * 8) * C4 + nt * 16 + m15] = acc[r];
    }
    // ---- value projection (y only): (16 x 128) @ (128 x 128) ----
    if constexpr (!IS_X) {
#pragma unroll 2
        for (int nt = 0; nt < 8; ++nt) {
            v8f acc = {};
#pragma unroll
            for (int kb = 0; kb < 4; ++kb)
                acc = wmma_bf16(af[kb], ldfrag(wfrag, FRAG_V + kb * 8 + nt, lane), acc);
            float bias = b_v[nt * 16 + m15];
#pragma unroll
            for (int r = 0; r < 8; ++r)
                yv[(size_t)(rowbase + r + half * 8) * CC + nt * 16 + m15] = acc[r] + bias;
        }
    }
}

__global__ void __launch_bounds__(256)
sna_linear(const float* __restrict__ x, const float* __restrict__ y,
           const bf16* __restrict__ wfrag,
           const float* __restrict__ b_lin, const float* __restrict__ b_v,
           float* __restrict__ xf, float* __restrict__ xq,
           float* __restrict__ yk, float* __restrict__ yv) {
    __shared__ bf16 in_lds[WAVES][16 * IN_LDSW];
    __shared__ bf16 act_lds[WAVES][16 * LDSW_BF];
    const int wave = threadIdx.x >> 5;
    const int lane = threadIdx.x & 31;
    const int tile_id = blockIdx.x * WAVES + wave;
    if (tile_id >= TILES) return;           // no barriers in this kernel
    const int rowbase = tile_id * 16;
    linear_tile<true>(x, wfrag, b_lin, b_v, xf, xq, yv, in_lds[wave], act_lds[wave],
                      rowbase, lane);
    linear_tile<false>(y, wfrag, b_lin, b_v, xf, yk, yv, in_lds[wave], act_lds[wave],
                       rowbase, lane);
}

// ---------------- kernel 2: energy + softmax + global attn sum ----------------
__global__ void __launch_bounds__(256)
sna_attn(const float* __restrict__ xq, const float* __restrict__ yk,
         const int* __restrict__ nbr, float* __restrict__ attn,
         float* __restrict__ attn_sum_g) {
    __shared__ float s_sum[TT * KK];
    if (threadIdx.x < TT * KK) s_sum[threadIdx.x] = 0.f;
    __syncthreads();

    int idx = blockIdx.x * blockDim.x + threadIdx.x;
    if (idx < NT) {
        int n = idx >> 2, t = idx & 3;
        float q[C4];
        const float* qp = xq + (size_t)idx * C4;
#pragma unroll
        for (int i = 0; i < C4; ++i) q[i] = qp[i];

        const int* nb = nbr + (size_t)n * KK;
        float e[KK];
        float mx = -3.4e38f;
#pragma unroll 2
        for (int k = 0; k < KK; ++k) {
            int nn = nb[k];
            const float* kp = yk + ((size_t)nn * TT + t) * C4;
            if (k + 4 < KK)   // pull upcoming gather line toward the WGP (L2-resident)
                __builtin_prefetch(yk + ((size_t)nb[k + 4] * TT + t) * C4, 0, 3);
            float s = 0.f;
#pragma unroll
            for (int i = 0; i < C4; ++i) s += q[i] * kp[i];
            e[k] = s;
            mx = fmaxf(mx, s);
        }
        float s = 0.f;
#pragma unroll
        for (int k = 0; k < KK; ++k) { e[k] = __expf(e[k] - mx); s += e[k]; }
        float inv = 1.f / s;
#pragma unroll
        for (int k = 0; k < KK; ++k) {
            float a = e[k] * inv;
            attn[(size_t)idx * KK + k] = a;
            atomicAdd(&s_sum[t * KK + k], a);       // ds_add_f32
        }
    }
    __syncthreads();
    if (threadIdx.x < TT * KK)
        atomicAdd(&attn_sum_g[threadIdx.x], s_sum[threadIdx.x]);
}

// ---------------- kernel 3: renorm, aggregate, trans-conv (WMMA), BN partials ----
__global__ void __launch_bounds__(256)
sna_apply(const float* __restrict__ xf, const float* __restrict__ yv,
          const float* __restrict__ attn, const float* __restrict__ attn_sum,
          const int* __restrict__ nbr, const bf16* __restrict__ wfrag,
          const float* __restrict__ b_t,
          float* __restrict__ z, float* __restrict__ bn_sum,
          float* __restrict__ bn_sum2) {
    __shared__ bf16 tiles[WAVES][16 * LDSW_BF];
    __shared__ float s_ainv[TT * KK];          // reciprocal of the global attn sums
    __shared__ float s_bn[2][CC];
    if (threadIdx.x < TT * KK)
        s_ainv[threadIdx.x] = 1.f / (1e-9f + attn_sum[threadIdx.x]);
    if (threadIdx.x < CC) { s_bn[0][threadIdx.x] = 0.f; s_bn[1][threadIdx.x] = 0.f; }
    __syncthreads();

    const int wave = threadIdx.x >> 5;
    const int lane = threadIdx.x & 31;
    const int tile_id = blockIdx.x * WAVES + wave;
    const int rowbase = tile_id * 16;

    if (tile_id < TILES) {
        bf16* t = tiles[wave];
        const int c0 = lane << 2;                  // 4 channels per lane
        // ---- y_r aggregation + diff = xf - y_r, packed bf16 into LDS ----
#pragma unroll 1
        for (int r = 0; r < 16; ++r) {
            int row = rowbase + r;
            int n = row >> 2, tt = row & 3;
            const int* nb = nbr + (size_t)n * KK;
            const float* ar = attn + (size_t)row * KK;
            float ax = 0.f, ay = 0.f, az = 0.f, aw = 0.f;
#pragma unroll 4
            for (int k = 0; k < KK; ++k) {
                int nn = nb[k];
                float w = ar[k] * s_ainv[tt * KK + k];
                const float4 v = *(const float4*)(yv + ((size_t)nn * TT + tt) * CC + c0);
                ax += w * v.x; ay += w * v.y; az += w * v.z; aw += w * v.w;
            }
            const float4 xv = *(const float4*)(xf + (size_t)row * CC + c0);
            *(unsigned int*)(t + r * LDSW_BF + c0)     = pack2bf(xv.x - ax, xv.y - ay);
            *(unsigned int*)(t + r * LDSW_BF + c0 + 2) = pack2bf(xv.z - az, xv.w - aw);
        }
        // ---- z = diff @ W_t^T + b_t  (16x128 @ 128x128) ----
        v16bf af[4];
#pragma unroll
        for (int kb = 0; kb < 4; ++kb) af[kb] = load_A_ldsbf(t, LDSW_BF, kb * 32, lane);
        const int m15 = lane & 15, half = lane >> 4;
#pragma unroll 2
        for (int nt = 0; nt < 8; ++nt) {
            v8f acc = {};
#pragma unroll
            for (int kb = 0; kb < 4; ++kb)
                acc = wmma_bf16(af[kb], ldfrag(wfrag, FRAG_T + kb * 8 + nt, lane), acc);
            int ch = nt * 16 + m15;
            float bias = b_t[ch];
            float s = 0.f, s2 = 0.f;
#pragma unroll
            for (int r = 0; r < 8; ++r) {
                float v = acc[r] + bias;
                z[(size_t)(rowbase + r + half * 8) * CC + ch] = v;
                s += v; s2 += v * v;
            }
            atomicAdd(&s_bn[0][ch], s);
            atomicAdd(&s_bn[1][ch], s2);
        }
    }
    __syncthreads();
    if (threadIdx.x < CC) {
        atomicAdd(&bn_sum[threadIdx.x], s_bn[0][threadIdx.x]);
        atomicAdd(&bn_sum2[threadIdx.x], s_bn[1][threadIdx.x]);
    }
}

// ---------------- kernel 4: BN finalize ----------------
__global__ void sna_bn(const float* __restrict__ bn_sum, const float* __restrict__ bn_sum2,
                       const float* __restrict__ gamma, const float* __restrict__ beta,
                       float* __restrict__ scale, float* __restrict__ shift) {
    int c = threadIdx.x;
    if (c < CC) {
        const float inv_n = 1.f / (float)NT;
        float mean = bn_sum[c] * inv_n;
        float var = bn_sum2[c] * inv_n - mean * mean;
        float is = rsqrtf(var + BN_EPS);
        float sc = gamma[c] * is;
        scale[c] = sc;
        shift[c] = beta[c] - mean * sc;
    }
}

// ---------------- kernel 5: residual + ReLU + max over T ----------------
__global__ void __launch_bounds__(256)
sna_out(const float* __restrict__ xf, const float* __restrict__ z,
        const float* __restrict__ scale, const float* __restrict__ shift,
        float* __restrict__ out) {
    int idx = blockIdx.x * blockDim.x + threadIdx.x;
    if (idx >= NN * CC) return;
    int n = idx / CC, c = idx & (CC - 1);
    float sc = scale[c], sh = shift[c];
    float m = -3.4e38f;
#pragma unroll
    for (int t = 0; t < TT; ++t) {
        size_t r = ((size_t)n * TT + t) * CC + c;
        float zn = z[r] * sc + sh;
        float v = xf[r] + fmaxf(zn, 0.f);
        m = fmaxf(m, v);
    }
    out[idx] = m;   // (N, C, 1) flat
}

// ---------------- host: launch pipeline ----------------
extern "C" void kernel_launch(void* const* d_in, const int* in_sizes, int n_in,
                              void* d_out, int out_size, void* d_ws, size_t ws_size,
                              hipStream_t stream) {
    const float* x     = (const float*)d_in[0];
    const float* y     = (const float*)d_in[1];
    const int*   nbr   = (const int*)d_in[2];
    const float* W_lin = (const float*)d_in[3];
    const float* b_lin = (const float*)d_in[4];
    const float* W_qk  = (const float*)d_in[5];
    const float* W_v   = (const float*)d_in[6];
    const float* b_v   = (const float*)d_in[7];
    const float* W_t   = (const float*)d_in[8];
    const float* b_t   = (const float*)d_in[9];
    const float* gamma = (const float*)d_in[10];
    const float* beta  = (const float*)d_in[11];

    float* ws = (float*)d_ws;
    float* xf   = ws;
    float* xq   = xf + (size_t)NT * CC;
    float* yk   = xq + (size_t)NT * C4;
    float* yv   = yk + (size_t)NT * C4;
    float* attn = yv + (size_t)NT * CC;
    float* z    = attn + (size_t)NT * KK;
    float* stats    = z + (size_t)NT * CC;   // [attn_sum(64)|bn_sum(128)|bn_sum2(128)|scale(128)|shift(128)]
    float* attn_sum = stats;
    float* bn_sum   = stats + 64;
    float* bn_sum2  = stats + 192;
    float* scale    = stats + 320;
    float* shift    = stats + 448;
    bf16*  wfrag    = (bf16*)(stats + 576);  // 88KB of packed bf16 weight fragments

    const int blocks_tile = (TILES + WAVES - 1) / WAVES;
    sna_zero<<<1, 320, 0, stream>>>(stats, 320);
    sna_pack<<<NFRAG, 32, 0, stream>>>(W_lin, W_qk, W_v, W_t, wfrag);
    sna_linear<<<blocks_tile, 256, 0, stream>>>(x, y, wfrag, b_lin, b_v,
                                                xf, xq, yk, yv);
    sna_attn<<<(NT + 255) / 256, 256, 0, stream>>>(xq, yk, nbr, attn, attn_sum);
    sna_apply<<<blocks_tile, 256, 0, stream>>>(xf, yv, attn, attn_sum, nbr, wfrag, b_t,
                                               z, bn_sum, bn_sum2);
    sna_bn<<<1, 128, 0, stream>>>(bn_sum, bn_sum2, gamma, beta, scale, shift);
    sna_out<<<(NN * CC + 255) / 256, 256, 0, stream>>>(xf, z, scale, shift, (float*)d_out);
}
